// GNNLayer_53541062312420
// MI455X (gfx1250) — matrix operand order repaired and checked by reading, not compiled, not run on hardware
//
#include <hip/hip_runtime.h>
#include <hip/hip_bf16.h>
#include <math.h>

typedef _Float16 v16h __attribute__((ext_vector_type(16)));
typedef float    v8f  __attribute__((ext_vector_type(8)));

// ---------------------------------------------------------------------------
// Small projections: Pr[r] = rela[r] @ Wr^T  (R x 64)
//                    Pq[b] = rela[q_rel[b]] @ Wqr^T + Wqr_b  (B x 64)
// ---------------------------------------------------------------------------
__global__ void proj_small_kernel(const float* __restrict__ rela,
                                  const int*   __restrict__ q_rel,
                                  const float* __restrict__ Wr,
                                  const float* __restrict__ Wqr,
                                  const float* __restrict__ Wqr_b,
                                  float* __restrict__ Pr,
                                  float* __restrict__ Pq,
                                  int R, int Bq) {
  const int row = blockIdx.x;
  const int a   = threadIdx.x;  // 0..63 output dim
  if (row < R) {
    const float* x = rela + (size_t)row * 64;
    const float* w = Wr   + (size_t)a   * 64;
    float s = 0.f;
    #pragma unroll
    for (int k = 0; k < 64; ++k) s += x[k] * w[k];
    Pr[(size_t)row * 64 + a] = s;
  } else {
    const int b = row - R;
    if (b < Bq) {
      const float* x = rela + (size_t)q_rel[b] * 64;
      const float* w = Wqr  + (size_t)a        * 64;
      float s = 0.f;
      #pragma unroll
      for (int k = 0; k < 64; ++k) s += x[k] * w[k];
      Pq[(size_t)b * 64 + a] = s + Wqr_b[a];
    }
  }
}

// ---------------------------------------------------------------------------
// Out(N x 64) = X(N x 64, f32) @ W(64 x 64, f32)^T  via v_wmma_f32_16x16x32_f16
// One wave32 computes one 16x16 output tile; K=64 -> two WMMAs.
// ---------------------------------------------------------------------------
__global__ void gemm_xwt_wmma_kernel(const float* __restrict__ X,
                                     const float* __restrict__ W,
                                     float* __restrict__ Out,
                                     int N) {
  const int lane   = threadIdx.x & 31;
  const int wave   = threadIdx.x >> 5;
  const int gwave  = blockIdx.x * (blockDim.x >> 5) + wave;
  const int tile_m = gwave >> 2;          // 4 col-tiles cover 64 output cols
  const int tile_n = gwave & 3;
  const int r0 = tile_m * 16;
  if (r0 >= N) return;
  const int c0  = tile_n * 16;
  const int hi  = (lane >> 4) & 1;
  const int l15 = lane & 15;

  // B fragments: B[k][n] = W[c0+n][k]
  v16h b0, b1;
  {
    const float* wrow = W + (size_t)(c0 + l15) * 64;
    #pragma unroll
    for (int h = 0; h < 16; ++h) {
      const int k = hi * 16 + h;
      b0[h] = (_Float16)wrow[k];
      b1[h] = (_Float16)wrow[k + 32];
    }
  }
  // A fragments from X rows
  v16h a0, a1;
  {
    int m = r0 + l15;
    if (m >= N) m = N - 1;   // clamp; clamped rows are never stored
    const float* xrow = X + (size_t)m * 64;
    #pragma unroll
    for (int h = 0; h < 16; ++h) {
      const int k = hi * 16 + h;
      a0[h] = (_Float16)xrow[k];
      a1[h] = (_Float16)xrow[k + 32];
    }
  }

  v8f c = {};
  c = __builtin_amdgcn_wmma_f32_16x16x32_f16(false, a0, false, b0, (short)0, c, false, false);
  c = __builtin_amdgcn_wmma_f32_16x16x32_f16(false, a1, false, b1, (short)0, c, false, false);

  // Epilogue: guard is wave-uniform -> single branch, unguarded fast path.
  float* out0 = Out + (size_t)(r0 + hi * 8) * 64 + c0 + l15;
  if (r0 + 16 <= N) {
    #pragma unroll
    for (int j = 0; j < 8; ++j) out0[(size_t)j * 64] = c[j];
  } else {
    #pragma unroll
    for (int j = 0; j < 8; ++j) {
      if (r0 + hi * 8 + j < N) out0[(size_t)j * 64] = c[j];
    }
  }
}

// ---------------------------------------------------------------------------
// Edge kernel: one wave32 per edge, 2 feature dims per lane.
// ---------------------------------------------------------------------------
__global__ void edge_kernel(const int*   __restrict__ edges,   // E x 6
                            const float* __restrict__ Ps,      // N x 64
                            const float* __restrict__ Pr,      // R x 64
                            const float* __restrict__ Pq,      // B x 64
                            const float* __restrict__ hidden,  // N x 64
                            const float* __restrict__ rela,    // R x 64
                            const float* __restrict__ waW,     // 64
                            const float* __restrict__ wab,     // 1
                            float* __restrict__ agg,           // N x 64
                            int E) {
  const int lane = threadIdx.x & 31;
  const int wave = threadIdx.x >> 5;
  const int e    = blockIdx.x * (blockDim.x >> 5) + wave;
  if (e >= E) return;

  const int r_idx = edges[(size_t)e * 6 + 0];
  const int rel   = edges[(size_t)e * 6 + 2];
  const int sub   = edges[(size_t)e * 6 + 4];
  const int obj   = edges[(size_t)e * 6 + 5];
  const int d     = lane * 2;

  const float2 ps = *(const float2*)(Ps  + (size_t)sub   * 64 + d);
  const float2 pr = *(const float2*)(Pr  + (size_t)rel   * 64 + d);
  const float2 pq = *(const float2*)(Pq  + (size_t)r_idx * 64 + d);
  const float2 wa = *(const float2*)(waW + d);

  float ax = ps.x + pr.x + pq.x; ax = ax > 0.f ? ax : 0.f;
  float ay = ps.y + pr.y + pq.y; ay = ay > 0.f ? ay : 0.f;
  float s = ax * wa.x + ay * wa.y;
  #pragma unroll
  for (int off = 16; off; off >>= 1) s += __shfl_xor(s, off, 32);
  const float alpha = 1.f / (1.f + __expf(-(s + wab[0])));

  const float2 hs = *(const float2*)(hidden + (size_t)sub * 64 + d);
  const float2 hr = *(const float2*)(rela   + (size_t)rel * 64 + d);
  float* p = agg + (size_t)obj * 64 + d;
  __hip_atomic_fetch_add(p + 0, alpha * (hs.x + hr.x),
                         __ATOMIC_RELAXED, __HIP_MEMORY_SCOPE_AGENT);
  __hip_atomic_fetch_add(p + 1, alpha * (hs.y + hr.y),
                         __ATOMIC_RELAXED, __HIP_MEMORY_SCOPE_AGENT);
}

// ---------------------------------------------------------------------------
extern "C" void kernel_launch(void* const* d_in, const int* in_sizes, int n_in,
                              void* d_out, int out_size, void* d_ws, size_t ws_size,
                              hipStream_t stream) {
  (void)n_in; (void)out_size; (void)ws_size;
  const int*   q_rel  = (const int*)  d_in[1];
  const float* hidden = (const float*)d_in[2];
  const int*   edges  = (const int*)  d_in[3];
  const float* rela   = (const float*)d_in[7];
  const float* Ws     = (const float*)d_in[8];
  const float* Wr     = (const float*)d_in[9];
  const float* WqrW   = (const float*)d_in[10];
  const float* Wqrb   = (const float*)d_in[11];
  const float* waW    = (const float*)d_in[12];
  const float* wab    = (const float*)d_in[13];
  const float* Wh     = (const float*)d_in[14];

  const int N  = in_sizes[2] / 64;   // 200000
  const int E  = in_sizes[3] / 6;    // 1000000
  const int R  = in_sizes[7] / 64;   // 401
  const int Bq = in_sizes[1];        // 8

  // Workspace layout (f32): Ps[N*64] | agg[N*64] | Pr[R*64] | Pq[Bq*64]
  float* Ps  = (float*)d_ws;
  float* agg = Ps  + (size_t)N * 64;
  float* Pr  = agg + (size_t)N * 64;
  float* Pq  = Pr  + (size_t)R * 64;

  hipMemsetAsync(agg, 0, (size_t)N * 64 * sizeof(float), stream);

  proj_small_kernel<<<R + Bq, 64, 0, stream>>>(rela, q_rel, Wr, WqrW, Wqrb,
                                               Pr, Pq, R, Bq);

  const int tiles  = ((N + 15) / 16) * 4;   // 16-row tiles x 4 col tiles
  const int blocks = (tiles + 7) / 8;       // 8 waves per 256-thread block
  gemm_xwt_wmma_kernel<<<blocks, 256, 0, stream>>>(hidden, Ws, Ps, N);

  const int eblocks = (E + 7) / 8;          // one wave per edge
  edge_kernel<<<eblocks, 256, 0, stream>>>(edges, Ps, Pr, Pq, hidden, rela,
                                           waW, wab, agg, E);

  gemm_xwt_wmma_kernel<<<blocks, 256, 0, stream>>>(agg, Wh, (float*)d_out, N);
}
